// FusedFP8Linear_40364102648219
// MI455X (gfx1250) — 8-bit matrix operand order repaired and checked by reading, compiled, not run on hardware
//
#include <hip/hip_runtime.h>
#include <stdint.h>

#define FP8_MAX 448.0f
#define EPS 1e-12f

typedef int   v16i __attribute__((ext_vector_type(16)));
typedef float v8f  __attribute__((ext_vector_type(8)));

__device__ __forceinline__ unsigned char f32_to_e4m3(float x) {
    unsigned int xb  = __float_as_uint(x);
    unsigned int sgn = (xb >> 24) & 0x80u;
    unsigned int ax  = xb & 0x7FFFFFFFu;
    unsigned char r;
    if (ax >= 0x43E00000u) {
        r = 0x7Eu;
    } else if (ax < 0x3C800000u) {
        float t = __uint_as_float(ax) * 512.0f;
        int m = (int)rintf(t);
        r = (unsigned char)m;
    } else {
        unsigned int drop = ax & 0xFFFFFu;
        unsigned int base = ax >> 20;
        base += (drop > 0x80000u) || ((drop == 0x80000u) && (base & 1u));
        int e = (int)((base >> 3) & 0xFFu) - 120;
        r = (unsigned char)((e << 3) | (base & 7u));
    }
    return (unsigned char)(r | sgn);
}

__global__ void init_kernel(unsigned int* slots) {
    slots[0] = 0u;
    slots[1] = 0u;
}

__global__ __launch_bounds__(256)
void amax_kernel(const float* __restrict__ src, long n4, unsigned int* __restrict__ slot) {
    long i      = (long)blockIdx.x * blockDim.x + threadIdx.x;
    long stride = (long)gridDim.x * blockDim.x;
    const float4* s4 = (const float4*)src;
    float m = 0.0f;
    for (long j = i; j < n4; j += stride) {
        float4 v = s4[j];
        m = fmaxf(m, fmaxf(fmaxf(fabsf(v.x), fabsf(v.y)),
                           fmaxf(fabsf(v.z), fabsf(v.w))));
    }
#pragma unroll
    for (int off = 16; off > 0; off >>= 1)
        m = fmaxf(m, __shfl_xor(m, off, 32));
    if ((threadIdx.x & 31) == 0)
        atomicMax(slot, __float_as_uint(m));
}

__global__ __launch_bounds__(256)
void quant_kernel(const float* __restrict__ src, unsigned int* __restrict__ dst,
                  long n4, const unsigned int* __restrict__ amax_slot) {
    float amax  = __uint_as_float(*amax_slot);
    float scale = FP8_MAX / fmaxf(amax, EPS);
    long i      = (long)blockIdx.x * blockDim.x + threadIdx.x;
    long stride = (long)gridDim.x * blockDim.x;
    const float4* s4 = (const float4*)src;
    for (long j = i; j < n4; j += stride) {
        float4 v = s4[j];
        float c0 = fminf(fmaxf(v.x * scale, -FP8_MAX), FP8_MAX);
        float c1 = fminf(fmaxf(v.y * scale, -FP8_MAX), FP8_MAX);
        float c2 = fminf(fmaxf(v.z * scale, -FP8_MAX), FP8_MAX);
        float c3 = fminf(fmaxf(v.w * scale, -FP8_MAX), FP8_MAX);
#if defined(__gfx1250__) && __has_builtin(__builtin_amdgcn_cvt_pk_fp8_f32)
        int p = __builtin_amdgcn_cvt_pk_fp8_f32(c0, c1, 0, false);
        p     = __builtin_amdgcn_cvt_pk_fp8_f32(c2, c3, p, true);
        dst[j] = (unsigned int)p;
#else
        unsigned int p =  (unsigned int)f32_to_e4m3(c0)
                       | ((unsigned int)f32_to_e4m3(c1) << 8)
                       | ((unsigned int)f32_to_e4m3(c2) << 16)
                       | ((unsigned int)f32_to_e4m3(c3) << 24);
        dst[j] = p;
#endif
    }
}

__global__ __launch_bounds__(256)
void gemm_fp8_kernel(const unsigned char* __restrict__ xq,
                     const unsigned char* __restrict__ wq,
                     const float* __restrict__ bias,
                     const unsigned int* __restrict__ amax_slots,
                     float* __restrict__ out,
                     int Din, int Dout) {
    const int lane = threadIdx.x & 31;
    const int wave = threadIdx.x >> 5;
    const int wm   = wave & 3;
    const int wn   = wave >> 2;
    const int h    = lane >> 4;
    const int l16  = lane & 15;

    const int rowBase = blockIdx.y * 128 + wm * 32;
    const int colBase = blockIdx.x * 128 + wn * 64;

    const unsigned char* aPtr[2];
    const unsigned char* bPtr[4];
#pragma unroll
    for (int mt = 0; mt < 2; ++mt)
        aPtr[mt] = xq + (size_t)(rowBase + 16 * mt + l16) * Din + 16 * h;
#pragma unroll
    for (int nt = 0; nt < 4; ++nt)
        bPtr[nt] = wq + (size_t)(colBase + 16 * nt + l16) * Din + 16 * h;

    v8f acc[2][4];
#pragma unroll
    for (int mt = 0; mt < 2; ++mt)
#pragma unroll
        for (int nt = 0; nt < 4; ++nt) {
            v8f z = {};
            acc[mt][nt] = z;
        }

    for (int k = 0; k < Din; k += 128) {
        v16i a[2];
        v16i b[4];
#pragma unroll
        for (int mt = 0; mt < 2; ++mt) {
#pragma unroll
            for (int bb = 0; bb < 2; ++bb)
#pragma unroll
                for (int g = 0; g < 4; ++g) {
                    uint2 d = *(const uint2*)(aPtr[mt] + k + 64 * bb + 32 * (g >> 1) + 8 * (g & 1));
                    a[mt][bb * 8 + g * 2 + 0] = (int)d.x;
                    a[mt][bb * 8 + g * 2 + 1] = (int)d.y;
                }
        }
#pragma unroll
        for (int nt = 0; nt < 4; ++nt) {
#pragma unroll
            for (int q = 0; q < 4; ++q) {
                uint4 d = *(const uint4*)(bPtr[nt] + k + 32 * q);
                b[nt][q * 4 + 0] = (int)d.x;
                b[nt][q * 4 + 1] = (int)d.y;
                b[nt][q * 4 + 2] = (int)d.z;
                b[nt][q * 4 + 3] = (int)d.w;
            }
        }
#pragma unroll
        for (int mt = 0; mt < 2; ++mt)
#pragma unroll
            for (int nt = 0; nt < 4; ++nt)
                acc[mt][nt] = __builtin_amdgcn_wmma_f32_16x16x128_fp8_fp8(
                    a[mt], b[nt], (short)0, acc[mt][nt], false, false);
    }

    float axm = fmaxf(__uint_as_float(amax_slots[0]), EPS);
    float awm = fmaxf(__uint_as_float(amax_slots[1]), EPS);
    float deq = (axm / FP8_MAX) * (awm / FP8_MAX);

#pragma unroll
    for (int mt = 0; mt < 2; ++mt) {
#pragma unroll
        for (int nt = 0; nt < 4; ++nt) {
            int col  = colBase + 16 * nt + l16;
            float bv = bias[col];
#pragma unroll
            for (int r = 0; r < 8; ++r) {
                int row = rowBase + 16 * mt + r + 8 * h;
                out[(size_t)row * Dout + col] = acc[mt][nt][r] * deq + bv;
            }
        }
    }
}

extern "C" void kernel_launch(void* const* d_in, const int* in_sizes, int n_in,
                              void* d_out, int out_size, void* d_ws, size_t ws_size,
                              hipStream_t stream) {
    const float* x    = (const float*)d_in[0];
    const float* w    = (const float*)d_in[1];
    const float* bias = (const float*)d_in[2];
    float* out        = (float*)d_out;

    long nx  = (long)in_sizes[0];
    long nw  = (long)in_sizes[1];
    int Dout = in_sizes[2];
    int Din  = (int)(nw / Dout);
    int T    = (int)(nx / Din);

    unsigned int*  amax = (unsigned int*)d_ws;
    unsigned char* xq   = (unsigned char*)d_ws + 256;
    unsigned char* wq   = xq + (size_t)nx;

    init_kernel<<<1, 1, 0, stream>>>(amax);

    amax_kernel<<<1024, 256, 0, stream>>>(x, nx >> 2, amax + 0);
    amax_kernel<<<1024, 256, 0, stream>>>(w, nw >> 2, amax + 1);

    quant_kernel<<<4096, 256, 0, stream>>>(x, (unsigned int*)xq, nx >> 2, amax + 0);
    quant_kernel<<<4096, 256, 0, stream>>>(w, (unsigned int*)wq, nw >> 2, amax + 1);

    dim3 grid(Dout / 128, T / 128);
    gemm_fp8_kernel<<<grid, 256, 0, stream>>>(xq, wq, bias, amax, out, Din, Dout);
}
